// MultiHeadedAttention_66898410602570
// MI455X (gfx1250) — matrix operand order repaired and checked by reading, not compiled, not run on hardware
//
#include <hip/hip_runtime.h>

// MI455X / gfx1250 MHA: bf16 WMMA projections + flash-attention core,
// with Tensor Data Mover (tensor_load_to_lds) staging for bf16 tiles.
// Workspace layout (64 MB): qh[16MB] kh[16MB] vhT[16MB] ctx[16MB], all bf16.

typedef __bf16 bf16;
typedef __attribute__((ext_vector_type(16))) __bf16 v16bf;
typedef __attribute__((ext_vector_type(8)))  float  v8f;
typedef __attribute__((ext_vector_type(4)))  unsigned int v4u;
typedef __attribute__((ext_vector_type(8)))  unsigned int v8u;

#define S_LEN 2048
#define D_DIM 1024
#define H_NUM 16
#define DK_DIM 64

union Frag32B { v16bf v; uint4 q[2]; };

// A-matrix fragment (16x32 bf16): lane holds row m=lane&15; 16 contiguous K
// halves 2*kbase..2*kbase+15 (kbase = 0 for lanes 0-15, 8 for 16-31).
__device__ __forceinline__ v16bf load_frag_a(const bf16* row, int kbase) {
  Frag32B f;
  f.q[0] = *(const uint4*)(row + 2 * kbase);
  f.q[1] = *(const uint4*)(row + 2 * kbase + 8);
  return f.v;
}

// B-matrix fragment (32x16 bf16): lane holds col n=lane&15; 16 contiguous K
// halves starting at the supplied pointer (caller adds +16 halves for lanes>=16).
__device__ __forceinline__ v16bf load_frag_b(const bf16* p) {
  Frag32B f;
  f.q[0] = *(const uint4*)(p);
  f.q[1] = *(const uint4*)(p + 8);
  return f.v;
}

__device__ __forceinline__ v8f wmma_bf16(v16bf a, v16bf b, v8f c) {
  return __builtin_amdgcn_wmma_f32_16x16x32_bf16(
      /*neg_a=*/false, a, /*neg_b=*/false, b,
      /*c_mod=*/(short)0, c, /*reuse_a=*/false, /*reuse_b=*/false);
}

// ---------------------------------------------------------------------------
// Tensor Data Mover: 2D bf16 tile load Global->LDS with row padding that
// reproduces the 72-half (144 B) skewed LDS row stride:
//   pad_interval=4 (pad every 32 DWORDs = 128 B row) ; pad_amount=3 (4 DWORDs).
// D# per ISA ch.8: group0 = {count=1, lds_addr, global_addr[56:0], type=2},
// group1 = {data_size=2B, pad cfg, tensor dims/strides, tile dims}.
// Issued by one wave; EXEC is ignored by TDM; completion via TENSORcnt.
// ---------------------------------------------------------------------------
__device__ __forceinline__ void tdm_load_tile_2d(
    unsigned int lds_off, unsigned long long gaddr, unsigned int tensor_dim0,
    unsigned int tensor_dim1, unsigned int tile_dim0, unsigned int tile_dim1,
    unsigned int stride0) {
  v4u g0;
  g0[0] = 1u;                                   // count=1, user descriptor
  g0[1] = lds_off;                              // lds_addr (bytes)
  g0[2] = (unsigned int)gaddr;                  // global_addr[31:0]
  g0[3] = (unsigned int)((gaddr >> 32) & 0x1FFFFFFull) | (2u << 30);  // type=2
  v8u g1;
  g1[0] = (1u << 16)       // data_size = 2 bytes
        | (1u << 20)       // pad_enable
        | (4u << 22)       // pad_interval: every 32 DWORDs (128 B)
        | (3u << 25);      // pad_amount: 4 DWORDs (16 B)
  g1[1] = (tensor_dim0 & 0xFFFFu) << 16;        // tensor_dim0[15:0] @ bits 63:48
  g1[2] = (tensor_dim0 >> 16) | ((tensor_dim1 & 0xFFFFu) << 16);
  g1[3] = (tensor_dim1 >> 16) | (tile_dim0 << 16);
  g1[4] = tile_dim1;                            // tile_dim1; tile_dim2=0
  g1[5] = stride0;                              // tensor_dim0_stride[31:0]
  g1[6] = 0u;                                   // stride0 hi, stride1 lo
  g1[7] = 0u;
  asm volatile("tensor_load_to_lds %0, %1" : : "s"(g0), "s"(g1) : "memory");
}

// ---------------------------------------------------------------------------
// Kernel 1: Y = X(f32)[M,1024] @ W^T(f32)[1024,1024] + bias -> bf16
//   transposed==0 : head-split  out[((b*16+h)*2048+s)*64 + dk]
//   transposed==1 : per-head Vt out[((b*16+h)*64+dk)*2048 + s]
// Block: 256 threads (8 waves), 64x64 output tile, K-step 64 (4 WMMA/stage).
// ---------------------------------------------------------------------------
__global__ __launch_bounds__(256) void proj_kernel(
    const float* __restrict__ X, const float* __restrict__ W,
    const float* __restrict__ bias, bf16* __restrict__ out, int transposed) {
  __shared__ bf16 As[64][72];
  __shared__ bf16 Bs[64][72];

  const int tid  = threadIdx.x;
  const int m0   = blockIdx.x * 64;
  const int n0   = blockIdx.y * 64;
  const int wave = tid >> 5;
  const int lane = tid & 31;
  const int mi   = (wave & 3) * 16;       // sub-tile row offset
  const int nj   = (wave >> 2) * 32;      // sub-tile col offset (2 tiles/wave)
  const int lane_n = lane & 15;
  const int kbase  = (lane < 16) ? 0 : 8;
  const int kbsel  = (lane < 16) ? 0 : 16;

  v8f acc0 = {};
  v8f acc1 = {};

  for (int k0 = 0; k0 < D_DIM; k0 += 64) {
    // Cooperative load + f32->bf16 convert into LDS (64x64 for A and B).
    for (int c = tid; c < 1024; c += 256) {
      const int row = c >> 4;
      const int col = (c & 15) * 4;
      float4 xv = *(const float4*)(X + (size_t)(m0 + row) * D_DIM + k0 + col);
      bf16* da = &As[row][col];
      da[0] = (bf16)xv.x; da[1] = (bf16)xv.y; da[2] = (bf16)xv.z; da[3] = (bf16)xv.w;
      float4 wv = *(const float4*)(W + (size_t)(n0 + row) * D_DIM + k0 + col);
      bf16* db = &Bs[row][col];
      db[0] = (bf16)wv.x; db[1] = (bf16)wv.y; db[2] = (bf16)wv.z; db[3] = (bf16)wv.w;
      if (k0 + 64 < D_DIM) {
        __builtin_prefetch(X + (size_t)(m0 + row) * D_DIM + k0 + 64 + col, 0, 0);
        __builtin_prefetch(W + (size_t)(n0 + row) * D_DIM + k0 + 64 + col, 0, 0);
      }
    }
    __syncthreads();

#pragma unroll
    for (int kk = 0; kk < 64; kk += 32) {
      v16bf a  = load_frag_a(&As[mi + lane_n][kk], kbase);
      v16bf b0 = load_frag_b(&Bs[nj + lane_n][kk + kbsel]);
      v16bf b1 = load_frag_b(&Bs[nj + 16 + lane_n][kk + kbsel]);
      acc0 = wmma_bf16(a, b0, acc0);
      acc1 = wmma_bf16(a, b1, acc1);
    }
    __syncthreads();
  }

  // C/D layout: lane n = lane&15; rows m = r + 8*(lane>>4), r = VGPR index.
  const int rbase = (lane >> 4) * 8;
#pragma unroll
  for (int t = 0; t < 2; ++t) {
    v8f acc = t ? acc1 : acc0;
    const int nc = n0 + nj + t * 16 + lane_n;
    const float bv = bias[nc];
    const int hh = nc >> 6;
    const int dk = nc & 63;
#pragma unroll
    for (int r = 0; r < 8; ++r) {
      const int m   = m0 + mi + rbase + r;
      const int bb  = m >> 11;            // batch (S = 2048)
      const int ss  = m & 2047;           // sequence pos
      const float val = acc[r] + bv;
      size_t idx;
      if (!transposed)
        idx = ((size_t)((bb * H_NUM + hh) * S_LEN + ss)) * DK_DIM + dk;
      else
        idx = ((size_t)((bb * H_NUM + hh) * DK_DIM + dk)) * S_LEN + ss;
      out[idx] = (bf16)val;
    }
  }
}

// ---------------------------------------------------------------------------
// Kernel 2: flash attention per (query-block, head, batch).
// Block: 128 threads (4 waves), each wave owns a 16-row query tile.
// K/V tiles staged by the Tensor Data Mover (wave 0 issues, TENSORcnt waits),
// mask tile converted by VALU in parallel with the DMA.
// ---------------------------------------------------------------------------
__global__ __launch_bounds__(128) void attn_kernel(
    const bf16* __restrict__ qh, const bf16* __restrict__ kh,
    const bf16* __restrict__ vhT, const int* __restrict__ mask,
    bf16* __restrict__ ctx) {
  __shared__ bf16  Ks[64][72];        // [key_local][dk]   (TDM-padded rows)
  __shared__ bf16  Vs[64][72];        // [dk][key_local]   (TDM-padded rows)
  __shared__ float Madd[64][68];      // additive mask (-1e9 or 0)
  __shared__ bf16  Ps[4][16][72];     // per-wave P staging (C->A relayout)

  const int q0   = blockIdx.x * 64;
  const int h    = blockIdx.y;
  const int b    = blockIdx.z;
  const int tid  = threadIdx.x;
  const int wave = tid >> 5;
  const int lane = tid & 31;
  const int lane_n  = lane & 15;
  const int halfsel = lane >> 4;          // 0 or 1
  const int kbase   = halfsel ? 8 : 0;    // A-frag K base
  const int kbsel   = halfsel ? 16 : 0;   // B-frag K base

  const size_t headoff = (size_t)(b * H_NUM + h) * S_LEN * DK_DIM;
  const bf16* qhead = qh + headoff;
  const bf16* khead = kh + headoff;
  const bf16* vhead = vhT + headoff;
  const unsigned int ksOff = (unsigned int)(unsigned long long)&Ks[0][0];
  const unsigned int vsOff = (unsigned int)(unsigned long long)&Vs[0][0];

  // Q fragments for this wave's 16 rows, kept in registers for all key blocks.
  const int sq = q0 + wave * 16 + lane_n;
  const bf16* qrow = qhead + (size_t)sq * DK_DIM;
  v16bf aq0 = load_frag_a(qrow, kbase);        // dk 0..31
  v16bf aq1 = load_frag_a(qrow + 32, kbase);   // dk 32..63

  float Mrow[8], Lrow[8];
  v8f oacc[4] = {};
#pragma unroll
  for (int r = 0; r < 8; ++r) { Mrow[r] = -3.0e38f; Lrow[r] = 0.0f; }

  for (int kb0 = 0; kb0 < S_LEN; kb0 += 64) {
    // ---- TDM staging: K tile (64 keys x 64 dk) and V^T tile (64 dk x 64 s)
    if (wave == 0) {
      // K: tensor [S_LEN rows x DK_DIM], tile rows kb0..kb0+63, full dk width.
      tdm_load_tile_2d(ksOff,
                       (unsigned long long)(khead + (size_t)kb0 * DK_DIM),
                       DK_DIM, S_LEN, DK_DIM, 64, DK_DIM);
      // V^T: tensor [DK_DIM rows x S_LEN], tile cols kb0..kb0+63, all 64 rows.
      tdm_load_tile_2d(vsOff, (unsigned long long)(vhead + kb0), S_LEN, DK_DIM,
                       64, DK_DIM, S_LEN);
    }
    // ---- mask tile -> additive f32 (overlaps the tensor DMA) ----
    const int* mrow = mask + ((size_t)b * S_LEN + q0) * S_LEN + kb0;
    for (int c = tid; c < 1024; c += 128) {
      const int row = c >> 4;
      const int col = (c & 15) * 4;
      int4 mv = *(const int4*)(mrow + (size_t)row * S_LEN + col);
      float4 fv;
      fv.x = mv.x ? 0.0f : -1.0e9f;
      fv.y = mv.y ? 0.0f : -1.0e9f;
      fv.z = mv.z ? 0.0f : -1.0e9f;
      fv.w = mv.w ? 0.0f : -1.0e9f;
      *(float4*)&Madd[row][col] = fv;
    }
    if (wave == 0) __builtin_amdgcn_s_wait_tensorcnt(0);
    __syncthreads();

    // ---- scores: S = Q(16x64) @ K^T(64x64) ----
    v8f sf[4];
#pragma unroll
    for (int nt = 0; nt < 4; ++nt) {
      sf[nt] = (v8f){};
      const int kn = nt * 16 + lane_n;                 // key column for this lane
      v16bf bk0 = load_frag_b(&Ks[kn][kbsel]);         // dk 0..31
      v16bf bk1 = load_frag_b(&Ks[kn][32 + kbsel]);    // dk 32..63
      sf[nt] = wmma_bf16(aq0, bk0, sf[nt]);
      sf[nt] = wmma_bf16(aq1, bk1, sf[nt]);
    }

    // ---- scale + mask + online softmax ----
    float sv[4][8];
#pragma unroll
    for (int nt = 0; nt < 4; ++nt)
#pragma unroll
      for (int r = 0; r < 8; ++r) {
        const int qloc = wave * 16 + halfsel * 8 + r;
        sv[nt][r] = sf[nt][r] * 0.125f + Madd[qloc][nt * 16 + lane_n];
      }

    float rmax[8];
#pragma unroll
    for (int r = 0; r < 8; ++r) {
      float v0 = fmaxf(sv[0][r], sv[1][r]);
      float v1 = fmaxf(sv[2][r], sv[3][r]);
      rmax[r] = fmaxf(v0, v1);
    }
#pragma unroll
    for (int off = 1; off < 16; off <<= 1)
#pragma unroll
      for (int r = 0; r < 8; ++r)
        rmax[r] = fmaxf(rmax[r], __shfl_xor(rmax[r], off, 32));

    float alpha[8], lsum[8];
#pragma unroll
    for (int r = 0; r < 8; ++r) {
      const float mnew = fmaxf(Mrow[r], rmax[r]);
      alpha[r] = __expf(Mrow[r] - mnew);
      Mrow[r]  = mnew;
      lsum[r]  = 0.0f;
    }
#pragma unroll
    for (int nt = 0; nt < 4; ++nt)
#pragma unroll
      for (int r = 0; r < 8; ++r) {
        const float p = __expf(sv[nt][r] - Mrow[r]);
        sv[nt][r] = p;
        lsum[r] += p;
      }
#pragma unroll
    for (int off = 1; off < 16; off <<= 1)
#pragma unroll
      for (int r = 0; r < 8; ++r)
        lsum[r] += __shfl_xor(lsum[r], off, 32);
#pragma unroll
    for (int r = 0; r < 8; ++r)
      Lrow[r] = Lrow[r] * alpha[r] + lsum[r];
#pragma unroll
    for (int dt = 0; dt < 4; ++dt)
#pragma unroll
      for (int r = 0; r < 8; ++r)
        oacc[dt][r] *= alpha[r];

    // ---- P: C-layout -> A-layout via LDS (bf16) ----
#pragma unroll
    for (int nt = 0; nt < 4; ++nt)
#pragma unroll
      for (int r = 0; r < 8; ++r)
        Ps[wave][halfsel * 8 + r][nt * 16 + lane_n] = (bf16)sv[nt][r];
    __syncthreads();

    // ---- O += P(16x64) @ V(64x64) ----
#pragma unroll
    for (int kk = 0; kk < 64; kk += 32) {
      v16bf ap = load_frag_a(&Ps[wave][lane_n][kk], kbase);
#pragma unroll
      for (int dt = 0; dt < 4; ++dt) {
        v16bf bv = load_frag_b(&Vs[dt * 16 + lane_n][kk + kbsel]);
        oacc[dt] = wmma_bf16(ap, bv, oacc[dt]);
      }
    }
    __syncthreads();
  }

  // ---- normalize + merge-heads store: ctx[b, s, h*64 + dk] (bf16) ----
#pragma unroll
  for (int dt = 0; dt < 4; ++dt) {
    const int dcol = h * DK_DIM + dt * 16 + lane_n;
#pragma unroll
    for (int r = 0; r < 8; ++r) {
      const int ss = q0 + wave * 16 + halfsel * 8 + r;
      const float val = oacc[dt][r] / (Lrow[r] + 1.0e-30f);
      ctx[((size_t)(b * S_LEN + ss)) * D_DIM + dcol] = (bf16)val;
    }
  }
}

// ---------------------------------------------------------------------------
// Kernel 3: out = ctx(bf16)[M,1024] @ Wo^T(f32) + bo -> f32 [M,1024]
// K-step 64; ctx tile staged by TDM, W tile converted f32->bf16 by VALU.
// ---------------------------------------------------------------------------
__global__ __launch_bounds__(256) void outproj_kernel(
    const bf16* __restrict__ ctx, const float* __restrict__ W,
    const float* __restrict__ bias, float* __restrict__ out) {
  __shared__ bf16 As[64][72];
  __shared__ bf16 Bs[64][72];

  const int tid  = threadIdx.x;
  const int m0   = blockIdx.x * 64;
  const int n0   = blockIdx.y * 64;
  const int wave = tid >> 5;
  const int lane = tid & 31;
  const int mi   = (wave & 3) * 16;
  const int nj   = (wave >> 2) * 32;
  const int lane_n = lane & 15;
  const int kbase  = (lane < 16) ? 0 : 8;
  const int kbsel  = (lane < 16) ? 0 : 16;
  const unsigned int asOff = (unsigned int)(unsigned long long)&As[0][0];

  v8f acc0 = {};
  v8f acc1 = {};

  for (int k0 = 0; k0 < D_DIM; k0 += 64) {
    // TDM: ctx tile (64 rows x 64 halves) from the [8192 x 1024] bf16 matrix.
    if (wave == 0)
      tdm_load_tile_2d(asOff,
                       (unsigned long long)(ctx + (size_t)m0 * D_DIM + k0),
                       D_DIM, 8192, 64, 64, D_DIM);
    // W tile: f32 -> bf16 conversion by all 256 threads (overlaps the DMA).
    for (int c = tid; c < 1024; c += 256) {
      const int row = c >> 4;
      const int col = (c & 15) * 4;
      float4 wv = *(const float4*)(W + (size_t)(n0 + row) * D_DIM + k0 + col);
      bf16* db = &Bs[row][col];
      db[0] = (bf16)wv.x; db[1] = (bf16)wv.y; db[2] = (bf16)wv.z; db[3] = (bf16)wv.w;
      if (k0 + 64 < D_DIM)
        __builtin_prefetch(W + (size_t)(n0 + row) * D_DIM + k0 + 64 + col, 0, 0);
    }
    if (wave == 0) __builtin_amdgcn_s_wait_tensorcnt(0);
    __syncthreads();

#pragma unroll
    for (int kk = 0; kk < 64; kk += 32) {
      v16bf a  = load_frag_a(&As[mi + lane_n][kk], kbase);
      v16bf b0 = load_frag_b(&Bs[nj + lane_n][kk + kbsel]);
      v16bf b1 = load_frag_b(&Bs[nj + 16 + lane_n][kk + kbsel]);
      acc0 = wmma_bf16(a, b0, acc0);
      acc1 = wmma_bf16(a, b1, acc1);
    }
    __syncthreads();
  }

  const int rbase = (lane >> 4) * 8;
#pragma unroll
  for (int t = 0; t < 2; ++t) {
    v8f acc = t ? acc1 : acc0;
    const int nc = n0 + nj + t * 16 + lane_n;
    const float bv = bias[nc];
#pragma unroll
    for (int r = 0; r < 8; ++r) {
      const int m = m0 + mi + rbase + r;
      out[(size_t)m * D_DIM + nc] = acc[r] + bv;
    }
  }
}

// ---------------------------------------------------------------------------
extern "C" void kernel_launch(void* const* d_in, const int* in_sizes, int n_in,
                              void* d_out, int out_size, void* d_ws,
                              size_t ws_size, hipStream_t stream) {
  const float* q    = (const float*)d_in[0];
  const float* k    = (const float*)d_in[1];
  const float* v    = (const float*)d_in[2];
  const int*   mask = (const int*)d_in[3];
  const float* Wq   = (const float*)d_in[4];
  const float* bq   = (const float*)d_in[5];
  const float* Wo   = (const float*)d_in[6];
  const float* bo   = (const float*)d_in[7];
  float* out = (float*)d_out;

  const size_t SEG = (size_t)4 * S_LEN * D_DIM * sizeof(bf16);  // 16 MB
  char* ws   = (char*)d_ws;
  bf16* qh   = (bf16*)(ws + 0 * SEG);
  bf16* kh   = (bf16*)(ws + 1 * SEG);
  bf16* vhT  = (bf16*)(ws + 2 * SEG);
  bf16* ctxb = (bf16*)(ws + 3 * SEG);

  dim3 gproj(128, 16);  // M/64 x N/64
  proj_kernel<<<gproj, 256, 0, stream>>>(q, Wq, bq, qh, 0);
  proj_kernel<<<gproj, 256, 0, stream>>>(k, Wq, bq, kh, 0);
  proj_kernel<<<gproj, 256, 0, stream>>>(v, Wq, bq, vhT, 1);

  attn_kernel<<<dim3(S_LEN / 64, H_NUM, 4), 128, 0, stream>>>(qh, kh, vhT, mask,
                                                              ctxb);

  outproj_kernel<<<gproj, 256, 0, stream>>>(ctxb, Wo, bo, out);
}
